// EfficientCostVolumeLayer_45251775431046
// MI455X (gfx1250) — matrix operand order repaired and checked by reading, not compiled, not run on hardware
//
#include <hip/hip_runtime.h>
#include <hip/hip_bf16.h>

typedef __attribute__((ext_vector_type(16))) _Float16 v16h;
typedef __attribute__((ext_vector_type(8)))  float    v8f;

#define CV_B 8
#define CV_C 128
#define CV_H 160
#define CV_W 160
#define CV_HW (CV_H * CV_W)
#define CV_OC 80

// One wave (32 lanes) per workgroup. Each wave produces out[b, 0:80, h, w0:w0+16].
// D = A x B with A = feat1 (16 pixels x 128 ch, f16), B = feat2 shifted window
// (128 ch x 16 cols, f16), accumulated in f32 via v_wmma_f32_16x16x32_f16.
__global__ __launch_bounds__(32)
void cost_volume_wmma_kernel(const float* __restrict__ f1,
                             const float* __restrict__ f2,
                             float* __restrict__ out)
{
    // Padded to 17 dwords/row so rows r and r+8 land in different banks.
    __shared__ float sD[2][16][17];

    const int w0   = blockIdx.x * 16;
    const int h    = blockIdx.y;
    const int b    = blockIdx.z;
    const int lane = threadIdx.x;      // 0..31
    const int m    = lane & 15;        // pixel-in-tile (M) / column-in-tile (N)
    const int half = lane >> 4;        // lane half selects K sub-range

    const float* f1b = f1 + (size_t)b * CV_C * CV_HW;
    const float* f2b = f2 + (size_t)b * CV_C * CV_HW;

    // ---------------- A fill: feat1[b, :, h, w0+m], ISA A-layout ----------------
    // lane<16: elements 0..7 -> K 32s+0..7,  8..15 -> K 32s+16..23
    // lane>=16: elements 0..7 -> K 32s+8..15, 8..15 -> K 32s+24..31
    v16h av[4];
    {
        const int koff = half * 8;
        const float* aBase = f1b + (size_t)h * CV_W + (w0 + m);
        #pragma unroll
        for (int s = 0; s < 4; ++s) {
            // per-s base so all 16 loads use immediate byte offsets (< 8.3 MB)
            const float* aS = aBase + (size_t)(32 * s + koff) * CV_HW;
            float ta[16];
            #pragma unroll
            for (int e = 0; e < 16; ++e) {
                const int krel = (e < 8) ? e : (16 + (e - 8)); // + koff already in aS
                ta[e] = aS[(size_t)krel * CV_HW];
            }
            #pragma unroll
            for (int e = 0; e < 16; ++e) {
                av[s][e] = (_Float16)ta[e];
            }
        }
    }

    const int n  = m;          // B column within tile
    const int kb = half * 8;   // B K offset within each 32-wide K block

    const v8f vzero = {0.f, 0.f, 0.f, 0.f, 0.f, 0.f, 0.f, 0.f};

    #pragma unroll 1
    for (int i = 0; i < 9; ++i) {
        const int  row   = h + i - 4;
        const bool rowok = (row >= 0) && (row < CV_H);
        const int  rowc  = row < 0 ? 0 : (row >= CV_H ? CV_H - 1 : row);

        // Prefetch next row offset's feat2 slab (global_prefetch_b8 path).
        if (i < 8) {
            const int prow = h + i - 3;
            if (prow >= 0 && prow < CV_H) {
                const float* p = f2b + (size_t)prow * CV_W + w0 +
                                 (size_t)(lane * 4) * CV_HW;
                __builtin_prefetch(p, 0, 1);
            }
        }

        v8f acc[2] = {vzero, vzero};

        #pragma unroll
        for (int t = 0; t < 2; ++t) {
            const int   col   = w0 - 4 + 16 * t + n;
            const bool  ok    = rowok && (col >= 0) && (col < CV_W);
            const int   colc  = col < 0 ? 0 : (col >= CV_W ? CV_W - 1 : col);
            const float maskf = ok ? 1.0f : 0.0f;          // zero-padding
            const float* bBase = f2b + (size_t)rowc * CV_W + colc;
            #pragma unroll
            for (int s = 0; s < 4; ++s) {
                // per-(t,s) base: 16 unconditional loads w/ immediate offsets
                const float* bS = bBase + (size_t)(32 * s + kb) * CV_HW;
                float tb[16];
                #pragma unroll
                for (int e = 0; e < 16; ++e) {
                    tb[e] = bS[(size_t)((e < 8) ? e : (16 + (e - 8))) * CV_HW];
                }
                v16h bv;
                #pragma unroll
                for (int e = 0; e < 16; ++e) {
                    bv[e] = (_Float16)(tb[e] * maskf);
                }
                // D = A*B + C, f32 accumulate
                acc[t] = __builtin_amdgcn_wmma_f32_16x16x32_f16(
                    /*neg_a=*/false, av[s], /*neg_b=*/false, bv,
                    /*c_mod=*/(short)0, acc[t],
                    /*reuse_a=*/false, /*reuse_b=*/false);
            }
        }

        // ---- spill D tiles to LDS for diagonal-band extraction ----
        __syncthreads();
        #pragma unroll
        for (int t = 0; t < 2; ++t) {
            #pragma unroll
            for (int r = 0; r < 8; ++r) {
                sD[t][r + 8 * half][n] = acc[t][r];
            }
        }
        __syncthreads();

        // out[b, d(i,j), h, w0+mm] = D[tile][mm][mm+j-16*tile] / C
        for (int e = lane; e < 144; e += 32) {
            const int mm = e / 9;
            const int j  = e % 9;
            if (i == 4 && j == 4) continue;                // skip zero displacement
            const int d = i * 9 + j -
                          (((i > 4) || (i == 4 && j > 4)) ? 1 : 0);
            const int c01 = mm + j;                        // 0..23 across the two tiles
            const float v = sD[c01 >> 4][mm][c01 & 15];
            out[(((size_t)b * CV_OC + d) * CV_H + h) * CV_W + (w0 + mm)] =
                v * (1.0f / (float)CV_C);
        }
    }
}

extern "C" void kernel_launch(void* const* d_in, const int* in_sizes, int n_in,
                              void* d_out, int out_size, void* d_ws, size_t ws_size,
                              hipStream_t stream) {
    (void)in_sizes; (void)n_in; (void)d_ws; (void)ws_size; (void)out_size;
    const float* feat1 = (const float*)d_in[0];
    const float* feat2 = (const float*)d_in[1];
    float* out = (float*)d_out;

    dim3 grid(CV_W / 16, CV_H, CV_B);  // 10 x 160 x 8
    dim3 block(32);                     // one wave32 per workgroup
    cost_volume_wmma_kernel<<<grid, block, 0, stream>>>(feat1, feat2, out);
}
